// ErnieImageAttention_16123307229704
// MI455X (gfx1250) — matrix operand order repaired and checked by reading, not compiled, not run on hardware
//
#include <hip/hip_runtime.h>

typedef __bf16 bf16_t;
typedef __attribute__((ext_vector_type(16))) __bf16 v16bf;
typedef __attribute__((ext_vector_type(8)))  __bf16 v8bf;
typedef __attribute__((ext_vector_type(8)))  float  v8f;

#define B_SZ   2
#define SEQ    1024
#define DMODEL 3072
#define NHEADS 24
#define HDIM   128
#define MTOT   (B_SZ * SEQ)   // 2048

// ---------------------------------------------------------------------------
// CDNA5 primitives
// ---------------------------------------------------------------------------
__device__ __forceinline__ v8f wmma_bf16(v16bf a, v16bf b, v8f c) {
  return __builtin_amdgcn_wmma_f32_16x16x32_bf16(false, a, false, b, (short)0, c,
                                                 false, false);
}

__device__ __forceinline__ v8f vzero8() {
  v8f z;
#pragma unroll
  for (int i = 0; i < 8; ++i) z[i] = 0.0f;
  return z;
}

// Async 16B global->LDS copy from pre-formed addresses (ASYNCcnt-tracked).
__device__ __forceinline__ void async_copy16_reg(uint32_t lds_off,
                                                 uint64_t gaddr) {
  asm volatile("global_load_async_to_lds_b128 %0, %1, off"
               :: "v"(lds_off), "v"(gaddr) : "memory");
}

// Variant with compile-time INST_OFFSET: the 24-bit immediate is added to
// BOTH the LDS and the global address, so one base register covers a whole
// tile when the two strides match.
template <int IMM>
__device__ __forceinline__ void async_copy16_imm(uint32_t lds_off,
                                                 uint64_t gaddr) {
  asm volatile("global_load_async_to_lds_b128 %0, %1, off offset:%2"
               :: "v"(lds_off), "v"(gaddr), "i"(IMM) : "memory");
}

__device__ __forceinline__ void wait_async0() {
  asm volatile("s_wait_asynccnt 0x0" ::: "memory");
}

// A-matrix fragment (16x32, 16-bit): lane l row M=l&15;
// lanes 0-15 hold K {0..7, 16..23}, lanes 16-31 hold K {8..15, 24..31}.
__device__ __forceinline__ v16bf load_frag_a(const bf16_t* base, int ldm) {
  const int lane = threadIdx.x & 31;
  const bf16_t* p = base + (lane & 15) * ldm + ((lane & 16) ? 8 : 0);
  union { v16bf v; v8bf h[2]; } u;
  u.h[0] = *(const v8bf*)(p);
  u.h[1] = *(const v8bf*)(p + 16);
  return u.v;
}

// B-matrix fragment (32x16, 16-bit): lane l column N=l&15;
// lanes 0-15 hold K {0..7, 16..23}, lanes 16-31 hold K {8..15, 24..31}.
__device__ __forceinline__ v16bf load_frag_b(const bf16_t* base, int ldm) {
  const int lane = threadIdx.x & 31;
  const bf16_t* p = base + (lane & 15) * ldm + ((lane & 16) ? 8 : 0);
  union { v16bf v; v8bf h[2]; } u;
  u.h[0] = *(const v8bf*)(p);
  u.h[1] = *(const v8bf*)(p + 16);
  return u.v;
}

union BF4 { bf16_t h[4]; unsigned long long u; };

// ---------------------------------------------------------------------------
// Kernel 0: f32 -> bf16 conversion pre-pass (makes the whole weight set
// L2-resident in bf16 and enables raw async staging). 8 elems / thread.
// ---------------------------------------------------------------------------
__global__ __launch_bounds__(256) void cvt_bf16_kernel(
    const float* __restrict__ src, bf16_t* __restrict__ dst, int n8) {
  int i = blockIdx.x * 256 + threadIdx.x;
  if (i >= n8) return;
  const float4* s = (const float4*)src + (size_t)i * 2;
  float4 a = s[0], b = s[1];
  BF4 p0, p1;
  p0.h[0] = (bf16_t)a.x; p0.h[1] = (bf16_t)a.y;
  p0.h[2] = (bf16_t)a.z; p0.h[3] = (bf16_t)a.w;
  p1.h[0] = (bf16_t)b.x; p1.h[1] = (bf16_t)b.y;
  p1.h[2] = (bf16_t)b.z; p1.h[3] = (bf16_t)b.w;
  ulonglong2 o; o.x = p0.u; o.y = p1.u;
  *(ulonglong2*)(dst + (size_t)i * 8) = o;
}

// ---------------------------------------------------------------------------
// Kernel 1: QKV projection (bf16 x bf16 -> per-z epilogue)
// z=0 -> qraw bf16 [b][s][e]; z=1 -> kraw; z=2 -> vtb bf16 [b][h][d][s]
// Double-buffered LDS tiles via async copies; persistent address registers.
// ---------------------------------------------------------------------------
__global__ __launch_bounds__(256) void qkv_gemm_kernel(
    const bf16_t* __restrict__ hsb,   // [2048][3072]
    const bf16_t* __restrict__ wqkv,  // [3][3072][3072]
    bf16_t* __restrict__ qraw, bf16_t* __restrict__ kraw,
    bf16_t* __restrict__ vtb) {
  __shared__ bf16_t sA[2][128][32];
  __shared__ bf16_t sB[2][128][32];

  const int which = blockIdx.z;
  const int m0 = blockIdx.x * 128;
  const int n0 = blockIdx.y * 128;
  const int tid = threadIdx.x;
  const int lane = tid & 31;
  const int wave = tid >> 5;
  const int wm = wave & 3;
  const int wn = wave >> 2;

  const bf16_t* gA = hsb + (size_t)m0 * DMODEL;
  const bf16_t* gB =
      wqkv + (size_t)which * DMODEL * DMODEL + (size_t)n0 * DMODEL;

  // Per-thread staging chunks (16B each): c0 = tid, c1 = tid + 256.
  const int r0 = tid >> 2, c0 = (tid & 3) * 8;
  const int r1 = (tid + 256) >> 2, c1 = ((tid + 256) & 3) * 8;
  uint64_t gA0 = (uint64_t)(uintptr_t)(gA + (size_t)r0 * DMODEL + c0);
  uint64_t gA1 = (uint64_t)(uintptr_t)(gA + (size_t)r1 * DMODEL + c1);
  uint64_t gB0 = (uint64_t)(uintptr_t)(gB + (size_t)r0 * DMODEL + c0);
  uint64_t gB1 = (uint64_t)(uintptr_t)(gB + (size_t)r1 * DMODEL + c1);
  uint32_t lA0[2], lA1[2], lB0[2], lB1[2];
#pragma unroll
  for (int p = 0; p < 2; ++p) {
    lA0[p] = (uint32_t)(uintptr_t)&sA[p][r0][c0];
    lA1[p] = (uint32_t)(uintptr_t)&sA[p][r1][c1];
    lB0[p] = (uint32_t)(uintptr_t)&sB[p][r0][c0];
    lB1[p] = (uint32_t)(uintptr_t)&sB[p][r1][c1];
  }

  auto issue = [&](int p) {
    async_copy16_reg(lA0[p], gA0);
    async_copy16_reg(lA1[p], gA1);
    async_copy16_reg(lB0[p], gB0);
    async_copy16_reg(lB1[p], gB1);
    gA0 += 64; gA1 += 64; gB0 += 64; gB1 += 64;  // advance one K-step (32 bf16)
  };

  v8f acc[2][4];
#pragma unroll
  for (int m = 0; m < 2; ++m)
#pragma unroll
    for (int n = 0; n < 4; ++n) acc[m][n] = vzero8();

  issue(0);
  for (int k0 = 0; k0 < DMODEL; k0 += 32) {
    const int cur = (k0 >> 5) & 1;
    wait_async0();
    __syncthreads();
    if (k0 + 32 < DMODEL) issue(cur ^ 1);

    v16bf a0 = load_frag_a(&sA[cur][wm * 32][0], 32);
    v16bf a1 = load_frag_a(&sA[cur][wm * 32 + 16][0], 32);
#pragma unroll
    for (int n = 0; n < 4; ++n) {
      v16bf bn = load_frag_b(&sB[cur][wn * 64 + n * 16][0], 32);
      acc[0][n] = wmma_bf16(a0, bn, acc[0][n]);
      acc[1][n] = wmma_bf16(a1, bn, acc[1][n]);
    }
  }

  const int nlo = lane & 15;
  const int rsel = (lane & 16) ? 8 : 0;
  if (which < 2) {
    bf16_t* C = (which == 0) ? qraw : kraw;
#pragma unroll
    for (int m = 0; m < 2; ++m)
#pragma unroll
      for (int n = 0; n < 4; ++n)
#pragma unroll
        for (int v = 0; v < 8; ++v) {
          int r = m0 + wm * 32 + m * 16 + v + rsel;
          int c = n0 + wn * 64 + n * 16 + nlo;
          C[(size_t)r * DMODEL + c] = (bf16_t)acc[m][n][v];
        }
  } else {  // V: bf16 transposed [b][h][d][s]
#pragma unroll
    for (int m = 0; m < 2; ++m)
#pragma unroll
      for (int n = 0; n < 4; ++n)
#pragma unroll
        for (int v = 0; v < 8; ++v) {
          int r = m0 + wm * 32 + m * 16 + v + rsel;
          int e = n0 + wn * 64 + n * 16 + nlo;
          int b = r >> 10, s = r & 1023;
          int h = e >> 7, d = e & 127;
          vtb[(((size_t)b * NHEADS + h) * HDIM + d) * SEQ + s] =
              (bf16_t)acc[m][n][v];
        }
  }
}

// ---------------------------------------------------------------------------
// Kernel 2: RMS norm + rotary + relayout to [b][h][s][d] (f32 math)
// ---------------------------------------------------------------------------
__global__ __launch_bounds__(128) void norm_rope_kernel(
    const bf16_t* __restrict__ qraw, const bf16_t* __restrict__ kraw,
    const float* __restrict__ rot, const float* __restrict__ nqw,
    const float* __restrict__ nkw, bf16_t* __restrict__ qb,
    bf16_t* __restrict__ kb) {
  const int idx = blockIdx.x;  // ((b*SEQ+s)*NHEADS + h)
  const int h = idx % NHEADS;
  const int s = (idx / NHEADS) % SEQ;
  const int b = idx / (NHEADS * SEQ);
  const int d = threadIdx.x;

  const bool is_q = (blockIdx.y == 0);
  const bf16_t* src = (is_q ? qraw : kraw) + (size_t)idx * HDIM;
  const float x = (float)src[d];

  __shared__ float red[128];
  __shared__ float xs[128];
  red[d] = x * x;
  __syncthreads();
#pragma unroll
  for (int off = 64; off > 0; off >>= 1) {
    if (d < off) red[d] += red[d + off];
    __syncthreads();
  }
  const float inv = rsqrtf(red[0] * (1.0f / 128.0f) + 1e-6f);
  const float* w = is_q ? nqw : nkw;
  const float xn = x * inv * w[d];
  xs[d] = xn;
  __syncthreads();

  const float fr = rot[(size_t)s * HDIM + d];
  const float c = cosf(fr), sn = sinf(fr);
  const float rv = (d < 64) ? -xs[d + 64] : xs[d - 64];
  const float out = xn * c + rv * sn;

  bf16_t* dst =
      (is_q ? qb : kb) + ((((size_t)b * NHEADS + h) * SEQ + s) * HDIM + d);
  *dst = (bf16_t)out;
}

// ---------------------------------------------------------------------------
// Kernel 3: flash attention per (b,h); async K/V^T staging into LDS.
// K block uses one base address + INST_OFFSET immediates (global and LDS
// strides match at 2048B); V^T keeps per-chunk addresses.
// ---------------------------------------------------------------------------
__global__ __launch_bounds__(128) void attn_kernel(
    const bf16_t* __restrict__ qb,   // [b][h][s][d]
    const bf16_t* __restrict__ kb,   // [b][h][s][d]
    const bf16_t* __restrict__ vtb,  // [b][h][d][s]
    bf16_t* __restrict__ ob) {       // [b][s][h*d]
  const int bh = blockIdx.y;
  const int q0 = blockIdx.x * 64;
  const int tid = threadIdx.x;
  const int lane = tid & 31;
  const int wave = tid >> 5;

  const bf16_t* Q = qb + (size_t)bh * SEQ * HDIM;
  const bf16_t* Kp = kb + (size_t)bh * SEQ * HDIM;
  const bf16_t* Vt = vtb + (size_t)bh * HDIM * SEQ;

  __shared__ bf16_t sK[64][128];
  __shared__ bf16_t sVT[128][64];
  __shared__ bf16_t sP[4][16][64];

  // Persistent staging addresses.
  uint64_t gk = (uint64_t)(uintptr_t)Kp + (uint64_t)tid * 16;
  const uint32_t lk = (uint32_t)(uintptr_t)&sK[0][0] + (uint32_t)tid * 16;
  uint64_t gv =
      (uint64_t)(uintptr_t)(Vt + (size_t)(tid >> 3) * SEQ + (tid & 7) * 8);
  const uint32_t lv = (uint32_t)(uintptr_t)&sVT[tid >> 3][(tid & 7) * 8];

  v16bf qa[4];
  {
    const int row = q0 + wave * 16 + (lane & 15);
    const bf16_t* p = Q + (size_t)row * HDIM + ((lane & 16) ? 8 : 0);
#pragma unroll
    for (int ds_ = 0; ds_ < 4; ++ds_) {
      union { v16bf v; v8bf h[2]; } u;
      u.h[0] = *(const v8bf*)(p + ds_ * 32);
      u.h[1] = *(const v8bf*)(p + ds_ * 32 + 16);
      qa[ds_] = u.v;
    }
  }

  v8f o[8];
#pragma unroll
  for (int n = 0; n < 8; ++n) o[n] = vzero8();
  float m_run[8], l_run[8];
#pragma unroll
  for (int v = 0; v < 8; ++v) { m_run[v] = -1.0e30f; l_run[v] = 0.0f; }

  const float scale = 0.08838834764831845f;  // 1/sqrt(128)
  const int rsel = (lane & 16) ? 8 : 0;

  for (int j0 = 0; j0 < SEQ; j0 += 64) {
    // ---- async stage K (64x128) and V^T (128x64) ----
    async_copy16_imm<0 * 2048>(lk, gk);
    async_copy16_imm<1 * 2048>(lk, gk);
    async_copy16_imm<2 * 2048>(lk, gk);
    async_copy16_imm<3 * 2048>(lk, gk);
    async_copy16_imm<4 * 2048>(lk, gk);
    async_copy16_imm<5 * 2048>(lk, gk);
    async_copy16_imm<6 * 2048>(lk, gk);
    async_copy16_imm<7 * 2048>(lk, gk);
#pragma unroll
    for (int i = 0; i < 8; ++i)
      async_copy16_reg(lv + i * 2048, gv + (uint64_t)i * 32768);
    wait_async0();
    __syncthreads();

    // ---- S = Q K^T (16q x 64k per wave) ----
    v8f sacc[4];
#pragma unroll
    for (int n = 0; n < 4; ++n) sacc[n] = vzero8();
#pragma unroll
    for (int dstep = 0; dstep < 4; ++dstep) {
#pragma unroll
      for (int n = 0; n < 4; ++n) {
        v16bf bk = load_frag_b(&sK[n * 16][dstep * 32], 128);
        sacc[n] = wmma_bf16(qa[dstep], bk, sacc[n]);
      }
    }

    // ---- online softmax (rows live in v + 8*(lane>=16)) ----
    float rmax[8];
#pragma unroll
    for (int v = 0; v < 8; ++v) rmax[v] = -1.0e30f;
#pragma unroll
    for (int n = 0; n < 4; ++n)
#pragma unroll
      for (int v = 0; v < 8; ++v) {
        sacc[n][v] *= scale;
        rmax[v] = fmaxf(rmax[v], sacc[n][v]);
      }
#pragma unroll
    for (int off = 8; off > 0; off >>= 1)
#pragma unroll
      for (int v = 0; v < 8; ++v)
        rmax[v] = fmaxf(rmax[v], __shfl_xor(rmax[v], off, 32));

    float fsc[8];
#pragma unroll
    for (int v = 0; v < 8; ++v) {
      float mn = fmaxf(m_run[v], rmax[v]);
      fsc[v] = __expf(m_run[v] - mn);
      m_run[v] = mn;
    }
#pragma unroll
    for (int n = 0; n < 8; ++n)
#pragma unroll
      for (int v = 0; v < 8; ++v) o[n][v] *= fsc[v];

    float rsum[8];
#pragma unroll
    for (int v = 0; v < 8; ++v) rsum[v] = 0.0f;
#pragma unroll
    for (int n = 0; n < 4; ++n)
#pragma unroll
      for (int v = 0; v < 8; ++v) {
        float p = __expf(sacc[n][v] - m_run[v]);
        sacc[n][v] = p;
        rsum[v] += p;
      }
#pragma unroll
    for (int off = 8; off > 0; off >>= 1)
#pragma unroll
      for (int v = 0; v < 8; ++v) rsum[v] += __shfl_xor(rsum[v], off, 32);
#pragma unroll
    for (int v = 0; v < 8; ++v) l_run[v] = l_run[v] * fsc[v] + rsum[v];

    // ---- P -> LDS (C-layout -> A-layout) ----
#pragma unroll
    for (int n = 0; n < 4; ++n)
#pragma unroll
      for (int v = 0; v < 8; ++v)
        sP[wave][v + rsel][n * 16 + (lane & 15)] = (bf16_t)sacc[n][v];

    // ---- O += P V ----
#pragma unroll
    for (int kstep = 0; kstep < 2; ++kstep) {
      v16bf pf = load_frag_a(&sP[wave][0][kstep * 32], 64);
#pragma unroll
      for (int n = 0; n < 8; ++n) {
        v16bf vf = load_frag_b(&sVT[n * 16][kstep * 32], 64);
        o[n] = wmma_bf16(pf, vf, o[n]);
      }
    }
    __syncthreads();

    gk += 16384;  // next 64 key rows (64*128*2 B)
    gv += 128;    // next 64 key cols (64*2 B)
  }

  const int b = bh / NHEADS, h = bh % NHEADS;
#pragma unroll
  for (int n = 0; n < 8; ++n)
#pragma unroll
    for (int v = 0; v < 8; ++v) {
      float val = o[n][v] / l_run[v];
      int s = q0 + wave * 16 + v + rsel;
      int d = n * 16 + (lane & 15);
      ob[((size_t)(b * SEQ + s)) * DMODEL + h * HDIM + d] = (bf16_t)val;
    }
}

// ---------------------------------------------------------------------------
// Kernel 4: output projection (bf16 x bf16 -> f32 d_out)
// ---------------------------------------------------------------------------
__global__ __launch_bounds__(256) void out_gemm_kernel(
    const bf16_t* __restrict__ Ab, const bf16_t* __restrict__ Wob,
    float* __restrict__ Cout) {
  __shared__ bf16_t sA[2][128][32];
  __shared__ bf16_t sB[2][128][32];

  const int m0 = blockIdx.x * 128;
  const int n0 = blockIdx.y * 128;
  const int tid = threadIdx.x;
  const int lane = tid & 31;
  const int wave = tid >> 5;
  const int wm = wave & 3;
  const int wn = wave >> 2;

  const bf16_t* gA = Ab + (size_t)m0 * DMODEL;
  const bf16_t* gB = Wob + (size_t)n0 * DMODEL;

  const int r0 = tid >> 2, c0 = (tid & 3) * 8;
  const int r1 = (tid + 256) >> 2, c1 = ((tid + 256) & 3) * 8;
  uint64_t gA0 = (uint64_t)(uintptr_t)(gA + (size_t)r0 * DMODEL + c0);
  uint64_t gA1 = (uint64_t)(uintptr_t)(gA + (size_t)r1 * DMODEL + c1);
  uint64_t gB0 = (uint64_t)(uintptr_t)(gB + (size_t)r0 * DMODEL + c0);
  uint64_t gB1 = (uint64_t)(uintptr_t)(gB + (size_t)r1 * DMODEL + c1);
  uint32_t lA0[2], lA1[2], lB0[2], lB1[2];
#pragma unroll
  for (int p = 0; p < 2; ++p) {
    lA0[p] = (uint32_t)(uintptr_t)&sA[p][r0][c0];
    lA1[p] = (uint32_t)(uintptr_t)&sA[p][r1][c1];
    lB0[p] = (uint32_t)(uintptr_t)&sB[p][r0][c0];
    lB1[p] = (uint32_t)(uintptr_t)&sB[p][r1][c1];
  }

  auto issue = [&](int p) {
    async_copy16_reg(lA0[p], gA0);
    async_copy16_reg(lA1[p], gA1);
    async_copy16_reg(lB0[p], gB0);
    async_copy16_reg(lB1[p], gB1);
    gA0 += 64; gA1 += 64; gB0 += 64; gB1 += 64;
  };

  v8f acc[2][4];
#pragma unroll
  for (int m = 0; m < 2; ++m)
#pragma unroll
    for (int n = 0; n < 4; ++n) acc[m][n] = vzero8();

  issue(0);
  for (int k0 = 0; k0 < DMODEL; k0 += 32) {
    const int cur = (k0 >> 5) & 1;
    wait_async0();
    __syncthreads();
    if (k0 + 32 < DMODEL) issue(cur ^ 1);

    v16bf a0 = load_frag_a(&sA[cur][wm * 32][0], 32);
    v16bf a1 = load_frag_a(&sA[cur][wm * 32 + 16][0], 32);
#pragma unroll
    for (int n = 0; n < 4; ++n) {
      v16bf bn = load_frag_b(&sB[cur][wn * 64 + n * 16][0], 32);
      acc[0][n] = wmma_bf16(a0, bn, acc[0][n]);
      acc[1][n] = wmma_bf16(a1, bn, acc[1][n]);
    }
  }

  const int nlo = lane & 15;
  const int rsel = (lane & 16) ? 8 : 0;
#pragma unroll
  for (int m = 0; m < 2; ++m)
#pragma unroll
    for (int n = 0; n < 4; ++n)
#pragma unroll
      for (int v = 0; v < 8; ++v) {
        int r = m0 + wm * 32 + m * 16 + v + rsel;
        int c = n0 + wn * 64 + n * 16 + nlo;
        Cout[(size_t)r * DMODEL + c] = acc[m][n][v];
      }
}

// ---------------------------------------------------------------------------
// Host launcher
// ---------------------------------------------------------------------------
extern "C" void kernel_launch(void* const* d_in, const int* in_sizes, int n_in,
                              void* d_out, int out_size, void* d_ws,
                              size_t ws_size, hipStream_t stream) {
  (void)in_sizes; (void)n_in; (void)out_size; (void)ws_size;
  const float* hs  = (const float*)d_in[0];
  const float* rot = (const float*)d_in[1];
  const float* Wq  = (const float*)d_in[2];
  const float* Wk  = (const float*)d_in[3];
  const float* Wv  = (const float*)d_in[4];
  const float* Wo  = (const float*)d_in[5];
  const float* nqw = (const float*)d_in[6];
  const float* nkw = (const float*)d_in[7];

  const size_t act = (size_t)MTOT * DMODEL;    // 6.29M elems
  const size_t wel = (size_t)DMODEL * DMODEL;  // 9.44M elems

  char* w = (char*)d_ws;
  bf16_t* hsb   = (bf16_t*)w; w += act * sizeof(bf16_t);
  bf16_t* wqkvb = (bf16_t*)w; w += 3 * wel * sizeof(bf16_t);
  bf16_t* wob   = (bf16_t*)w; w += wel * sizeof(bf16_t);
  bf16_t* qraw  = (bf16_t*)w; w += act * sizeof(bf16_t);
  bf16_t* kraw  = (bf16_t*)w; w += act * sizeof(bf16_t);
  bf16_t* qb    = (bf16_t*)w; w += act * sizeof(bf16_t);
  bf16_t* kb    = (bf16_t*)w; w += act * sizeof(bf16_t);
  bf16_t* vtb   = (bf16_t*)w; w += act * sizeof(bf16_t);
  bf16_t* ob    = (bf16_t*)w; w += act * sizeof(bf16_t);

  const int act8 = (int)(act / 8), wel8 = (int)(wel / 8);
  cvt_bf16_kernel<<<(act8 + 255) / 256, 256, 0, stream>>>(hs, hsb, act8);
  cvt_bf16_kernel<<<(wel8 + 255) / 256, 256, 0, stream>>>(Wq, wqkvb, wel8);
  cvt_bf16_kernel<<<(wel8 + 255) / 256, 256, 0, stream>>>(Wk, wqkvb + wel, wel8);
  cvt_bf16_kernel<<<(wel8 + 255) / 256, 256, 0, stream>>>(Wv, wqkvb + 2 * wel,
                                                          wel8);
  cvt_bf16_kernel<<<(wel8 + 255) / 256, 256, 0, stream>>>(Wo, wob, wel8);

  qkv_gemm_kernel<<<dim3(MTOT / 128, DMODEL / 128, 3), 256, 0, stream>>>(
      hsb, wqkvb, qraw, kraw, vtb);
  norm_rope_kernel<<<dim3(B_SZ * SEQ * NHEADS, 2), 128, 0, stream>>>(
      qraw, kraw, rot, nqw, nkw, qb, kb);
  attn_kernel<<<dim3(SEQ / 64, B_SZ * NHEADS), 128, 0, stream>>>(qb, kb, vtb,
                                                                 ob);
  out_gemm_kernel<<<dim3(MTOT / 128, DMODEL / 128), 256, 0, stream>>>(
      ob, wob, (float*)d_out);
}
